// Encoder_43155831390481
// MI455X (gfx1250) — compile-verified, matrix operand order checked by reading, not run on hardware
//
#include <hip/hip_runtime.h>
#include <hip/hip_bf16.h>
#include <stdint.h>
#include <stddef.h>

typedef __attribute__((ext_vector_type(16))) _Float16 v16h;
typedef __attribute__((ext_vector_type(8)))  _Float16 v8h;
typedef __attribute__((ext_vector_type(8)))  float    v8f;

// Device-scope relaxed fp32 atomic add -> native global_atomic_add_f32 on gfx1250.
__device__ __forceinline__ void atomAddF(float* p, float v) {
    __hip_atomic_fetch_add(p, v, __ATOMIC_RELAXED, __HIP_MEMORY_SCOPE_AGENT);
}

// ---------------- degree / normalization ----------------

__global__ void k_fill1(float* __restrict__ p, int n) {
    int i = blockIdx.x * blockDim.x + threadIdx.x;
    if (i < n) p[i] = 1.0f;  // self-loop contributes 1 to every degree
}

__global__ void k_deg(const long long* __restrict__ dst, float* __restrict__ deg, int E) {
    int e = blockIdx.x * blockDim.x + threadIdx.x;
    if (e < E) atomAddF(&deg[(int)dst[e]], 1.0f);
}

__global__ void k_rsqrt(float* __restrict__ p, int n) {
    int i = blockIdx.x * blockDim.x + threadIdx.x;
    if (i < n) p[i] = rsqrtf(p[i]);  // deg >= 1 always (self-loops)
}

__global__ void k_norm(const long long* __restrict__ src, const long long* __restrict__ dst,
                       const float* __restrict__ dinv, float* __restrict__ nrm, int E) {
    int e = blockIdx.x * blockDim.x + threadIdx.x;
    if (e < E) nrm[e] = dinv[(int)src[e]] * dinv[(int)dst[e]];
}

// ---------------- precision conversion / weight packing ----------------

__global__ void k_f32_to_f16(const float* __restrict__ x, _Float16* __restrict__ y, long n) {
    long i = (long)blockIdx.x * blockDim.x + threadIdx.x;
    if (i < n) y[i] = (_Float16)x[i];
}

// Pack row-major W[K x Ncol] (f32) into the CDNA5 WMMA B-fragment layout (f16):
// Bp[((kt*nNT + nt)*32 + lane)*16 + j] = W[kt*32 + g*8 + (j&7) + ((j>>3)<<4)][nt*16 + (lane&15)]
// with g = lane>>4  (16-bit B matrix 32x16 layout, ISA 7.12.2).
__global__ void k_packW(const float* __restrict__ W, _Float16* __restrict__ Bp, int K, int Ncol) {
    int nNT = Ncol >> 4;
    int nKT = K >> 5;
    int total = nKT * nNT * 32 * 16;
    int idx = blockIdx.x * blockDim.x + threadIdx.x;
    if (idx >= total) return;
    int j  = idx & 15;
    int L  = (idx >> 4) & 31;
    int t  = idx >> 9;
    int nt = t % nNT;
    int kt = t / nNT;
    int g  = L >> 4;
    int k  = kt * 32 + g * 8 + (j & 7) + ((j >> 3) << 4);
    int n  = nt * 16 + (L & 15);
    Bp[idx] = (_Float16)W[(size_t)k * Ncol + n];
}

// ---------------- WMMA GEMM: D[M x Ncol] = A[M x K] (f16) * Bpacked (f16) ----------------
// One wave per 16x16 output tile; 8 waves / block. Requires M % 16 == 0 (50000 = 3125*16).
__global__ void k_gemm_wmma(const _Float16* __restrict__ A, const _Float16* __restrict__ Bp,
                            float* __restrict__ D, int M, int K, int Ncol) {
    const int lane = threadIdx.x & 31;
    const int wave = threadIdx.x >> 5;
    const int nNT  = Ncol >> 4;
    const int mt   = blockIdx.x * (8 / nNT) + wave / nNT;
    const int nt   = wave % nNT;
    if (mt * 16 >= M) return;              // wave-uniform exit, EXEC stays all-1s for WMMA

    const int g = lane >> 4;
    const int r = lane & 15;
    // A fragment: lane holds row (lane&15), K groups {g*8..g*8+7} and {g*8+16..g*8+23}
    const _Float16* arow = A + (size_t)(mt * 16 + r) * K + g * 8;
    const int nKT = K >> 5;

    v8f acc = {};
    for (int kt = 0; kt < nKT; ++kt) {
        v8h a0 = *(const v8h*)(arow + kt * 32);        // K = base .. base+7
        v8h a1 = *(const v8h*)(arow + kt * 32 + 16);   // K = base+16 .. base+23
        __builtin_prefetch(arow + kt * 32 + 32, 0, 0); // speculative prefetch of next K-tile
        v16h a = __builtin_shufflevector(a0, a1, 0,1,2,3,4,5,6,7,8,9,10,11,12,13,14,15);
        v16h b = *(const v16h*)(Bp + (((size_t)kt * nNT + nt) * 32 + lane) * 16);
        acc = __builtin_amdgcn_wmma_f32_16x16x32_f16(
            /*neg_a=*/false, a, /*neg_b=*/false, b,
            /*c_mod=*/(short)0, acc, /*reuse_a=*/false, /*reuse_b=*/false);
    }

    // D fragment: lane holds col (lane&15); VGPR i -> row (lane>>4)*8 + i
    float* drow = D + (size_t)(mt * 16 + g * 8) * Ncol + nt * 16 + r;
    #pragma unroll
    for (int i = 0; i < 8; ++i) drow[(size_t)i * Ncol] = acc[i];
}

// ---------------- aggregation ----------------

// y[i][c] = bias[c] + h[i][c] * dinv[i]^2   (bias + self-loop message, no atomics)
__global__ void k_agg_init(const float* __restrict__ h, const float* __restrict__ dinv,
                           const float* __restrict__ bias, float* __restrict__ y,
                           int N, int cshift) {
    long idx = (long)blockIdx.x * blockDim.x + threadIdx.x;
    long total = (long)N << cshift;
    if (idx >= total) return;
    int i = (int)(idx >> cshift);
    int c = (int)(idx & ((1 << cshift) - 1));
    float di = dinv[i];
    y[idx] = bias[c] + h[idx] * di * di;
}

// y[dst[e]][c] += h[src[e]][c] * norm[e]
__global__ void k_agg_edge(const float* __restrict__ h, const long long* __restrict__ src,
                           const long long* __restrict__ dst, const float* __restrict__ nrm,
                           float* __restrict__ y, int E, int cshift) {
    long idx = (long)blockIdx.x * blockDim.x + threadIdx.x;
    long total = (long)E << cshift;
    if (idx >= total) return;
    int e = (int)(idx >> cshift);
    int c = (int)(idx & ((1 << cshift) - 1));
    int s = (int)src[e];
    int d = (int)dst[e];
    atomAddF(&y[((long)d << cshift) + c], h[((long)s << cshift) + c] * nrm[e]);
}

// ---------------- host side ----------------

static inline size_t align256(size_t v) { return (v + 255) & ~(size_t)255; }

extern "C" void kernel_launch(void* const* d_in, const int* in_sizes, int n_in,
                              void* d_out, int out_size, void* d_ws, size_t ws_size,
                              hipStream_t stream) {
    const float*     x   = (const float*)d_in[0];
    const long long* ei  = (const long long*)d_in[1];   // int64 [2, E]
    const float*     W1  = (const float*)d_in[2];
    const float*     b1  = (const float*)d_in[3];
    const float*     W2  = (const float*)d_in[4];
    const float*     b2  = (const float*)d_in[5];

    const int HID = in_sizes[3];            // 128
    const int OUT = in_sizes[5];            // 64
    const int FIN = in_sizes[2] / HID;      // 128
    const int N   = in_sizes[0] / FIN;      // 50000
    const int E   = in_sizes[1] / 2;        // 800000
    const long long* srcI = ei;
    const long long* dstI = ei + E;

    // workspace layout (~67.5 MB)
    char* ws = (char*)d_ws;
    size_t off = 0;
    float*    dinv = (float*)(ws + off);    off = align256(off + (size_t)N * 4);
    float*    nrm  = (float*)(ws + off);    off = align256(off + (size_t)E * 4);
    _Float16* w1p  = (_Float16*)(ws + off); off = align256(off + (size_t)FIN * HID * 2);
    _Float16* w2p  = (_Float16*)(ws + off); off = align256(off + (size_t)HID * OUT * 2);
    _Float16* xh   = (_Float16*)(ws + off); off = align256(off + (size_t)N * FIN * 2); // reused for y1 in f16
    float*    hbuf = (float*)(ws + off);    off = align256(off + (size_t)N * HID * 4); // reused for layer-2 GEMM out
    float*    y1   = (float*)(ws + off);    off = align256(off + (size_t)N * HID * 4);
    float*    outp = (float*)d_out;

    const dim3 blk(256);
    auto nb = [](long n) { return dim3((unsigned)((n + 255) / 256)); };

    // graph normalization (shared by both layers)
    k_fill1 <<<nb(N), blk, 0, stream>>>(dinv, N);
    k_deg   <<<nb(E), blk, 0, stream>>>(dstI, dinv, E);
    k_rsqrt <<<nb(N), blk, 0, stream>>>(dinv, N);
    k_norm  <<<nb(E), blk, 0, stream>>>(srcI, dstI, dinv, nrm, E);

    const int mtTiles = N / 16;   // 3125 (exact)

    // ---- layer 1 ----
    k_f32_to_f16 <<<nb((long)N * FIN), blk, 0, stream>>>(x, xh, (long)N * FIN);
    k_packW      <<<nb((FIN >> 5) * (HID >> 4) * 512), blk, 0, stream>>>(W1, w1p, FIN, HID);
    {
        int nNT = HID >> 4, mtPer = 8 / nNT;
        int blocks = (mtTiles + mtPer - 1) / mtPer;
        k_gemm_wmma <<<dim3(blocks), blk, 0, stream>>>(xh, w1p, hbuf, N, FIN, HID);
    }
    int cs1 = __builtin_ctz(HID);   // 7
    k_agg_init <<<nb((long)N * HID), blk, 0, stream>>>(hbuf, dinv, b1, y1, N, cs1);
    k_agg_edge <<<nb((long)E * HID), blk, 0, stream>>>(hbuf, srcI, dstI, nrm, y1, E, cs1);

    // ---- layer 2 ----
    k_f32_to_f16 <<<nb((long)N * HID), blk, 0, stream>>>(y1, xh, (long)N * HID);
    k_packW      <<<nb((HID >> 5) * (OUT >> 4) * 512), blk, 0, stream>>>(W2, w2p, HID, OUT);
    {
        int nNT = OUT >> 4, mtPer = 8 / nNT;
        int blocks = (mtTiles + mtPer - 1) / mtPer;
        k_gemm_wmma <<<dim3(blocks), blk, 0, stream>>>(xh, w2p, hbuf, N, HID, OUT);
    }
    int cs2 = __builtin_ctz(OUT);   // 6
    k_agg_init <<<nb((long)N * OUT), blk, 0, stream>>>(hbuf, dinv, b2, outp, N, cs2);
    k_agg_edge <<<nb((long)E * OUT), blk, 0, stream>>>(hbuf, srcI, dstI, nrm, outp, E, cs2);
}
